// Model_80513456931020
// MI455X (gfx1250) — matrix operand order repaired and checked by reading, not compiled, not run on hardware
//
#include <hip/hip_runtime.h>
#include <hip/hip_bf16.h>

typedef __attribute__((ext_vector_type(16))) _Float16 v16h;
typedef __attribute__((ext_vector_type(8)))  float    v8f;
typedef int v4i_b128 __attribute__((vector_size(16)));   // matches builtin param

#define N_NODES 50000
#define EDGE_DIM 54
#define HIDDEN 64
#define TILE_E 16     // layer-2 edges per wave-tile
#define TILE1 32      // layer-1 edges per block-tile (two 16-row M-tiles)
#define KPAD 64       // 54 attrs + 1 bias lane + pad -> 2 WMMA k-steps of 32

#define GLOBAL_AS __attribute__((address_space(1)))
#define LDS_AS    __attribute__((address_space(3)))

#if __has_builtin(__builtin_amdgcn_global_load_async_to_lds_b128) && \
    __has_builtin(__builtin_amdgcn_s_wait_asynccnt)
#define HAVE_ASYNC 1
#else
#define HAVE_ASYNC 0
#endif

// ---------------------------------------------------------------------------
// K1: agg[n][o] = x[n] @ root1[:,o] + bias1[o]   (baseline before scatter-add)
// ---------------------------------------------------------------------------
__global__ __launch_bounds__(256) void init_agg(const float* __restrict__ x,
                                                const float* __restrict__ root1,
                                                const float* __restrict__ bias1,
                                                float* __restrict__ agg) {
  int idx = blockIdx.x * blockDim.x + threadIdx.x;
  if (idx >= N_NODES * HIDDEN) return;
  int n = idx >> 6, o = idx & 63;
  float acc = bias1[o];
#pragma unroll
  for (int i = 0; i < 4; ++i) acc += x[n * 4 + i] * root1[i * HIDDEN + o];
  agg[idx] = acc;
}

// ---------------------------------------------------------------------------
// K2: layer-1 edge kernel.
// GEMM [32 edges x 64(k)] x [64 x 256] via v_wmma_f32_16x16x32_f16,
// msgs[e][o] = sum_i x[src[e]][i] * W[e][i*64+o], atomic scatter into agg.
// Wave w owns output columns [16w, 16w+16); edge_attr staged to LDS with
// double-buffered async LDS-DMA (ASYNCcnt) when available.
// ---------------------------------------------------------------------------
__global__ __launch_bounds__(128) void l1_edges(const float* __restrict__ x,
                                                const int* __restrict__ src,
                                                const int* __restrict__ dst,
                                                const float* __restrict__ edge_attr,
                                                const float* __restrict__ w_e1,
                                                const float* __restrict__ b_e1,
                                                float* __restrict__ agg,
                                                int n_tiles) {
  __shared__ _Float16 w_lds[KPAD][256];                    // 32 KB weights (f16)
  __shared__ __align__(16) float raw[2][TILE1 * EDGE_DIM]; // 13.5 KB raw f32 (dbl buf)
  __shared__ _Float16 a16[TILE1][KPAD];                    // 4 KB A panel (f16)
  __shared__ float    xj_lds[TILE1][4];
  __shared__ int      dst_lds[TILE1];

  const int tid  = threadIdx.x;
  const int lane = tid & 31;
  const int wave = tid >> 5;          // 0..3 = output column slice
  const int n    = lane & 15;
  const int hi   = (lane >> 4) & 1;
  const int m    = lane & 15;
  const int sel  = hi * 8;

#if HAVE_ASYNC
#define STAGE_TILE(T, BUF)                                                          \
  do {                                                                              \
    const float* gp_ = edge_attr + (size_t)(T) * (TILE1 * EDGE_DIM);                \
    GLOBAL_AS v4i_b128* g_ = (GLOBAL_AS v4i_b128*)gp_;                              \
    LDS_AS v4i_b128*    l_ = (LDS_AS v4i_b128*)&raw[BUF][0];                        \
    for (int c16 = tid; c16 < (TILE1 * EDGE_DIM) / 4; c16 += 128)                   \
      __builtin_amdgcn_global_load_async_to_lds_b128(g_ + c16, l_ + c16, 0, 0);     \
  } while (0)
#define STAGE_WAIT() __builtin_amdgcn_s_wait_asynccnt(0)
#else
#define STAGE_TILE(T, BUF)                                                          \
  do {                                                                              \
    const float* gp_ = edge_attr + (size_t)(T) * (TILE1 * EDGE_DIM);                \
    for (int c = tid; c < TILE1 * EDGE_DIM; c += 128) raw[BUF][c] = gp_[c];         \
  } while (0)
#define STAGE_WAIT() do { } while (0)
#endif

  // Stage weights (f32 -> f16) into LDS once per block; rows 0..53 = w_e1, 54 = b_e1.
  for (int idx = tid; idx < KPAD * 256; idx += 128) {
    int d = idx >> 8, c = idx & 255;
    float v = (d < EDGE_DIM) ? w_e1[d * 256 + c] : (d == EDGE_DIM ? b_e1[c] : 0.0f);
    w_lds[d][c] = (_Float16)v;
  }
  // Constant pad columns of the A panel: col 54 = 1.0 (bias), 55..63 = 0. Written once.
  for (int idx = tid; idx < TILE1 * (KPAD - EDGE_DIM); idx += 128) {
    int e = idx / (KPAD - EDGE_DIM), d = idx % (KPAD - EDGE_DIM);
    a16[e][EDGE_DIM + d] = (d == 0) ? (_Float16)1.0f : (_Float16)0.0f;
  }
  __syncthreads();

  // Persistent B fragments: (input ch i 0..3) x (kstep s 0..1) for this wave's slice.
  // B layout (32x16 f16): lane half hi, element e <-> K = 32s + 8*hi + e + (e>=8 ? 8 : 0), N = n.
  v16h B[8];
#pragma unroll
  for (int i = 0; i < 4; ++i) {
    const int col = i * 64 + wave * 16 + n;
#pragma unroll
    for (int s = 0; s < 2; ++s) {
      v16h b;
#pragma unroll
      for (int e = 0; e < 16; ++e) b[e] = w_lds[32 * s + 8 * hi + e + (e >= 8 ? 8 : 0)][col];
      B[i * 2 + s] = b;
    }
  }

  // Pipelined tile loop: batch issued in iteration i is waited at top of i+1.
  int tile = blockIdx.x;
  const int stride = gridDim.x;
  if (tile < n_tiles) STAGE_TILE(tile, 0);
  int p = 0;
  for (; tile < n_tiles; tile += stride, p ^= 1) {
    STAGE_WAIT();
    __syncthreads();                      // raw[p] visible; a16 free to overwrite
    int nxt = tile + stride;
    if (nxt < n_tiles) STAGE_TILE(nxt, p ^ 1);

    const int base = tile * TILE1;
    // Convert raw f32 -> f16 A panel (cols 0..53 only; pad cols are static).
    for (int idx = tid; idx < TILE1 * EDGE_DIM; idx += 128) {
      int e = idx / EDGE_DIM, d = idx - e * EDGE_DIM;
      a16[e][d] = (_Float16)raw[p][idx];
    }
    {
      int e = tid >> 2, i = tid & 3;      // 128 threads = 32 edges x 4 feats
      xj_lds[e][i] = x[(size_t)src[base + e] * 4 + i];
    }
    if (tid < TILE1) dst_lds[tid] = dst[base + tid];
    __syncthreads();

#pragma unroll
    for (int mt = 0; mt < 2; ++mt) {
      const int row = mt * 16 + m;
      // A fragments (16x32 f16): e<8 -> K=32s+sel+e ; e>=8 -> K=32s+16+sel+(e-8)
      v16h A0, A1;
#pragma unroll
      for (int e = 0; e < 8; ++e) {
        A0[e]     = a16[row][sel + e];
        A0[8 + e] = a16[row][16 + sel + e];
        A1[e]     = a16[row][32 + sel + e];
        A1[8 + e] = a16[row][48 + sel + e];
      }

      float msg[8] = {0.f, 0.f, 0.f, 0.f, 0.f, 0.f, 0.f, 0.f};
#pragma unroll
      for (int i = 0; i < 4; ++i) {
        v8f c = {};
        c = __builtin_amdgcn_wmma_f32_16x16x32_f16(false, A0, false, B[i * 2 + 0],
                                                   (short)0, c, false, false);
        c = __builtin_amdgcn_wmma_f32_16x16x32_f16(false, A1, false, B[i * 2 + 1],
                                                   (short)0, c, false, false);
#pragma unroll
        for (int r = 0; r < 8; ++r) {
          int mc = mt * 16 + (hi ? (r + 8) : r);   // C layout: VGPR r -> row r / r+8
          msg[r] += xj_lds[mc][i] * c[r];
        }
      }
#pragma unroll
      for (int r = 0; r < 8; ++r) {
        int mc = mt * 16 + (hi ? (r + 8) : r);
        atomicAdd(&agg[(size_t)dst_lds[mc] * HIDDEN + wave * 16 + n], msg[r]);
      }
    }
  }
#undef STAGE_TILE
#undef STAGE_WAIT
}

// ---------------------------------------------------------------------------
// K3: h = relu(agg) in place; out[n] = h[n] . root2 + bias2  (one warp/node)
// ---------------------------------------------------------------------------
__global__ __launch_bounds__(256) void relu_root2(float* __restrict__ hbuf,
                                                  const float* __restrict__ root2,
                                                  const float* __restrict__ bias2,
                                                  float* __restrict__ out) {
  int node = (blockIdx.x * blockDim.x + threadIdx.x) >> 5;
  int lane = threadIdx.x & 31;
  if (node >= N_NODES) return;
  float acc = 0.f;
#pragma unroll
  for (int k = lane; k < HIDDEN; k += 32) {
    float v = fmaxf(hbuf[(size_t)node * HIDDEN + k], 0.f);
    hbuf[(size_t)node * HIDDEN + k] = v;
    acc += v * root2[k];
  }
  acc += __shfl_xor(acc, 1);
  acc += __shfl_xor(acc, 2);
  acc += __shfl_xor(acc, 4);
  acc += __shfl_xor(acc, 8);
  acc += __shfl_xor(acc, 16);
  if (lane == 0) out[node] = acc + bias2[0];
}

// ---------------------------------------------------------------------------
// K4: layer-2 edge kernel. W2[e][k] = a[e]@w_e2 + b_e2 via WMMA (K pad 54->64,
// 4 col tiles), msgs2[e] = W2[e][:] . h[src[e]][:], scalar atomic into out.
// One wave per 16-edge tile, wave-private staging.
// ---------------------------------------------------------------------------
__global__ __launch_bounds__(128) void l2_edges(const float* __restrict__ h,
                                                const int* __restrict__ src,
                                                const int* __restrict__ dst,
                                                const float* __restrict__ edge_attr,
                                                const float* __restrict__ w_e2,
                                                const float* __restrict__ b_e2,
                                                float* __restrict__ out,
                                                int n_tiles) {
  __shared__ _Float16 w_lds[KPAD][HIDDEN];           // 8 KB
  __shared__ _Float16 a_lds[4][TILE_E][KPAD];        // 8 KB (per-wave staging)
  __shared__ int      src_lds[4][TILE_E];
  __shared__ int      dst_lds[4][TILE_E];

  const int tid  = threadIdx.x;
  const int lane = tid & 31;
  const int wave = tid >> 5;
  const int n    = lane & 15;
  const int hi   = (lane >> 4) & 1;
  const int m    = lane & 15;
  const int sel  = hi * 8;

  for (int idx = tid; idx < KPAD * HIDDEN; idx += blockDim.x) {
    int d = idx >> 6, c = idx & 63;
    float v = (d < EDGE_DIM) ? w_e2[d * HIDDEN + c] : (d == EDGE_DIM ? b_e2[c] : 0.0f);
    w_lds[d][c] = (_Float16)v;
  }
  __syncthreads();

  // Persistent B fragments: (col tile t 0..3) x (kstep s 0..1)
  v16h B[8];
#pragma unroll
  for (int t = 0; t < 4; ++t) {
#pragma unroll
    for (int s = 0; s < 2; ++s) {
      v16h b;
#pragma unroll
      for (int e = 0; e < 16; ++e) b[e] = w_lds[32 * s + 8 * hi + e + (e >= 8 ? 8 : 0)][t * 16 + n];
      B[t * 2 + s] = b;
    }
  }

  const int wstride = gridDim.x * 4;
  for (int tile = blockIdx.x * 4 + wave; tile < n_tiles; tile += wstride) {
    const int base = tile * TILE_E;
    for (int idx = lane; idx < TILE_E * KPAD; idx += 32) {
      int e = idx >> 6, d = idx & 63;
      float v = (d < EDGE_DIM) ? edge_attr[(size_t)(base + e) * EDGE_DIM + d]
                               : (d == EDGE_DIM ? 1.0f : 0.0f);
      a_lds[wave][e][d] = (_Float16)v;
    }
    if (lane < TILE_E) {
      src_lds[wave][lane] = src[base + lane];
      dst_lds[wave][lane] = dst[base + lane];
    }

    v16h A0, A1;
#pragma unroll
    for (int e = 0; e < 8; ++e) {
      A0[e]     = a_lds[wave][m][sel + e];
      A0[8 + e] = a_lds[wave][m][16 + sel + e];
      A1[e]     = a_lds[wave][m][32 + sel + e];
      A1[8 + e] = a_lds[wave][m][48 + sel + e];
    }

    float part[8] = {0.f, 0.f, 0.f, 0.f, 0.f, 0.f, 0.f, 0.f};
#pragma unroll
    for (int t = 0; t < 4; ++t) {
      v8f c = {};
      c = __builtin_amdgcn_wmma_f32_16x16x32_f16(false, A0, false, B[t * 2 + 0],
                                                 (short)0, c, false, false);
      c = __builtin_amdgcn_wmma_f32_16x16x32_f16(false, A1, false, B[t * 2 + 1],
                                                 (short)0, c, false, false);
#pragma unroll
      for (int r = 0; r < 8; ++r) {
        int mc = hi ? (r + 8) : r;
        part[r] += c[r] * h[(size_t)src_lds[wave][mc] * HIDDEN + t * 16 + n];
      }
    }
#pragma unroll
    for (int r = 0; r < 8; ++r) {
      float v = part[r];
      v += __shfl_xor(v, 1);
      v += __shfl_xor(v, 2);
      v += __shfl_xor(v, 4);
      v += __shfl_xor(v, 8);
      if (n == 0) {
        int mc = hi ? (r + 8) : r;
        atomicAdd(&out[dst_lds[wave][mc]], v);
      }
    }
  }
}

// ---------------------------------------------------------------------------
extern "C" void kernel_launch(void* const* d_in, const int* in_sizes, int n_in,
                              void* d_out, int out_size, void* d_ws, size_t ws_size,
                              hipStream_t stream) {
  const float* x      = (const float*)d_in[0];
  const int*   eidx   = (const int*)d_in[1];   // [2, E] int32
  const float* eattr  = (const float*)d_in[2];
  const float* w_e1   = (const float*)d_in[3];
  const float* b_e1   = (const float*)d_in[4];
  const float* root1  = (const float*)d_in[5];
  const float* bias1  = (const float*)d_in[6];
  const float* w_e2   = (const float*)d_in[7];
  const float* b_e2   = (const float*)d_in[8];
  const float* root2  = (const float*)d_in[9];
  const float* bias2  = (const float*)d_in[10];
  float* out  = (float*)d_out;
  float* hbuf = (float*)d_ws;                  // [N_NODES, 64] f32 (12.8 MB)

  const int E = in_sizes[1] / 2;
  const int* srcp = eidx;
  const int* dstp = eidx + E;

  init_agg<<<(N_NODES * HIDDEN + 255) / 256, 256, 0, stream>>>(x, root1, bias1, hbuf);
  l1_edges<<<1024, 128, 0, stream>>>(x, srcp, dstp, eattr, w_e1, b_e1, hbuf, E / TILE1);
  relu_root2<<<(N_NODES * 32 + 255) / 256, 256, 0, stream>>>(hbuf, root2, bias2, out);
  l2_edges<<<1024, 128, 0, stream>>>(hbuf, srcp, dstp, eattr, w_e2, b_e2, out, E / TILE_E);
}
